// EagleChunk3_36575941493066
// MI455X (gfx1250) — compile-verified, matrix operand order checked by reading, not run on hardware
//
#include <hip/hip_runtime.h>
#include <hip/hip_bf16.h>
#include <stdint.h>

typedef _Float16 f16;
typedef _Float16 v16h __attribute__((ext_vector_type(16)));
typedef _Float16 v8h  __attribute__((ext_vector_type(8)));
typedef float    v8f  __attribute__((ext_vector_type(8)));

namespace {
constexpr int S   = 1024;
constexpr int D   = 2048;
constexpr int H   = 8;
constexpr int KVH = 2;
constexpr int HD  = 256;
constexpr int F   = 4096;
constexpr int PLD = 256;
constexpr int L   = 8;
constexpr int LDSN = 40;            // padded half-stride of LDS tile rows
constexpr int TROWS = 128 + 256;    // A rows + B rows per buffer
constexpr uint32_t BUFBYTES = TROWS * LDSN * sizeof(f16);
}

// ---------------------------------------------------------------------------
// Async DMA: copy 16 bytes per lane from global to LDS (ASYNCcnt-tracked).
// ---------------------------------------------------------------------------
__device__ __forceinline__ void async_copy_b128(uint32_t lds_off, const f16* gaddr) {
  asm volatile("global_load_async_to_lds_b128 %0, %1, off"
               :: "v"(lds_off), "v"(gaddr) : "memory");
}
__device__ __forceinline__ void wait_async0() {
  asm volatile("s_wait_asynccnt 0x0" ::: "memory");
}

// ---------------------------------------------------------------------------
// Gather a 16x32 f16 WMMA operand fragment from an LDS tile stored K-contig
// per row (row = M for A, row = N for B), stride LDSN halves.
// Lane layout per ISA: lanes 0-15 hold K 0-7 & 16-23 of row (lane&15);
// lanes 16-31 hold K 8-15 & 24-31.  Two ds_load_b128 per fragment.
// ---------------------------------------------------------------------------
__device__ __forceinline__ v16h load_frag16(const f16* p) {
  v8h lo = *(const v8h*)(p);
  v8h hi = *(const v8h*)(p + 16);
  v16h r;
#pragma unroll
  for (int i = 0; i < 8; ++i) { r[i] = lo[i]; r[i + 8] = hi[i]; }
  return r;
}

// ---------------------------------------------------------------------------
// WMMA GEMM (all-f16 operands, A*B^T form):
//   C[z] = A[z] * B[z/bDiv]^T (+ Cin[z])
//   A: M x K f16, row stride lda (K contiguous)
//   B: N x K f16, row stride ldb (K contiguous)
//   Output: f32 (Cf) or f16 (Ch).
//   Block tile 128x256, 8 waves (2x4), each wave 64x64 = 4x4 fragments
//   -> 16 v_wmma_f32_16x16x32_f16 per K-step.
//   Double-buffered LDS staging via global_load_async_to_lds_b128: the DMA
//   for tile k+1 overlaps the fragment loads + WMMAs of tile k.
// ---------------------------------------------------------------------------
__global__ __launch_bounds__(256) void wmma_gemm_f16_kernel(
    const f16* __restrict__ A, const f16* __restrict__ B,
    const float* __restrict__ Cin, float* __restrict__ Cf, f16* __restrict__ Ch,
    int K, int lda, int ldb, int ldc, int ldcin,
    long long aHS, long long bHS, long long cHS, long long cinHS, int bDiv)
{
  __shared__ f16 lds[2 * TROWS * LDSN];   // [buf][A:128 | B:256 rows][LDSN]

  const int z = blockIdx.z;
  A += (long long)z * aHS;
  B += (long long)(z / bDiv) * bHS;
  if (Cf)  Cf  += (long long)z * cHS;
  if (Ch)  Ch  += (long long)z * cHS;
  if (Cin) Cin += (long long)z * cinHS;

  const int tid  = threadIdx.x;
  const int lane = tid & 31;
  const int wv   = tid >> 5;
  const int wm   = (wv >> 2) * 64;    // wave sub-tile row (2 groups)
  const int wn   = (wv & 3) * 64;     // wave sub-tile col (4 groups)
  const int m0   = blockIdx.y * 128;
  const int n0   = blockIdx.x * 256;

  const int r16 = lane & 15;
  const int kh  = (lane >> 4) * 8;    // K-half select (0 or 8)

  // ---- staging: A tile 128x32 = 512 chunks (2/thr), B tile 256x32 = 1024 (4/thr)
  // buffer-0 LDS byte offsets; ping-pong selected by adding a uniform delta.
  const f16* gA[2];
  const f16* gB[4];
  uint32_t lA0[2], lB0[4];
#pragma unroll
  for (int i = 0; i < 2; ++i) {
    const int c = tid + i * 256;
    const int r = c >> 2, cc = (c & 3) * 8;
    gA[i]  = A + (long long)(m0 + r) * lda + cc;
    lA0[i] = (uint32_t)(uintptr_t)(lds + r * LDSN + cc);
  }
#pragma unroll
  for (int j = 0; j < 4; ++j) {
    const int c = tid + j * 256;
    const int r = c >> 2, cc = (c & 3) * 8;
    gB[j]  = B + (long long)(n0 + r) * ldb + cc;
    lB0[j] = (uint32_t)(uintptr_t)(lds + (128 + r) * LDSN + cc);
  }

  auto issue = [&](uint32_t delta) {
#pragma unroll
    for (int i = 0; i < 2; ++i) { async_copy_b128(lA0[i] + delta, gA[i]); gA[i] += 32; }
#pragma unroll
    for (int j = 0; j < 4; ++j) { async_copy_b128(lB0[j] + delta, gB[j]); gB[j] += 32; }
  };

  v8f acc[4][4];
#pragma unroll
  for (int i = 0; i < 4; ++i)
#pragma unroll
    for (int j = 0; j < 4; ++j) acc[i][j] = (v8f){};

  const int nk = K >> 5;
  issue(0);
  for (int it = 0; it < nk; ++it) {
    wait_async0();
    __syncthreads();
    const int cur = it & 1;
    if (it + 1 < nk) issue(cur ? 0u : BUFBYTES);   // fill the other buffer

    const f16* Ab = lds + cur * (TROWS * LDSN);
    const f16* Bb = Ab + 128 * LDSN;
    v16h a[4], b[4];
#pragma unroll
    for (int i = 0; i < 4; ++i)
      a[i] = load_frag16(Ab + (wm + i * 16 + r16) * LDSN + kh);
#pragma unroll
    for (int j = 0; j < 4; ++j)
      b[j] = load_frag16(Bb + (wn + j * 16 + r16) * LDSN + kh);

#pragma unroll
    for (int i = 0; i < 4; ++i)
#pragma unroll
      for (int j = 0; j < 4; ++j)
        acc[i][j] = __builtin_amdgcn_wmma_f32_16x16x32_f16(
            false, a[i], false, b[j], (short)0, acc[i][j], false, false);
  }

  // epilogue: D fragment layout (VGPR r -> row rhi+r, col = lane&15)
  // 32-bit index math (all extents fit in int)
  const int cn  = lane & 15;
  const int rhi = (lane >> 4) * 8;
#pragma unroll
  for (int i = 0; i < 4; ++i) {
#pragma unroll
    for (int j = 0; j < 4; ++j) {
      const int col = n0 + wn + j * 16 + cn;
#pragma unroll
      for (int r = 0; r < 8; ++r) {
        const int row = m0 + wm + i * 16 + rhi + r;
        float v = acc[i][j][r];
        if (Cin) v += Cin[row * ldcin + col];
        if (Ch) Ch[row * ldc + col] = (f16)v;
        else    Cf[row * ldc + col] = v;
      }
    }
  }
}

// ---------------------------------------------------------------------------
// Tiled transpose + convert: out[z][c*R + r] = (f16) in[z][r*ldin + c]
// Grid (C/32, R/32, z), block 256 (32x8 logical).
// ---------------------------------------------------------------------------
__global__ __launch_bounds__(256) void transpose_cvt_f32_kernel(
    const float* __restrict__ in, f16* __restrict__ out,
    int R, int C, int ldin, long long inBS, long long outBS)
{
  __shared__ f16 tile[32][33];
  in  += (long long)blockIdx.z * inBS;
  out += (long long)blockIdx.z * outBS;
  const int r0 = blockIdx.y * 32, c0 = blockIdx.x * 32;
  const int tx = threadIdx.x & 31, ty = threadIdx.x >> 5;
#pragma unroll
  for (int j = 0; j < 4; ++j)
    tile[ty + j * 8][tx] = (f16)in[(long long)(r0 + ty + j * 8) * ldin + (c0 + tx)];
  __syncthreads();
#pragma unroll
  for (int j = 0; j < 4; ++j)
    out[(long long)(c0 + ty + j * 8) * R + (r0 + tx)] = tile[tx][ty + j * 8];
}

__global__ __launch_bounds__(256) void transpose_cvt_f16_kernel(
    const f16* __restrict__ in, f16* __restrict__ out,
    int R, int C, int ldin, long long inBS, long long outBS)
{
  __shared__ f16 tile[32][33];
  in  += (long long)blockIdx.z * inBS;
  out += (long long)blockIdx.z * outBS;
  const int r0 = blockIdx.y * 32, c0 = blockIdx.x * 32;
  const int tx = threadIdx.x & 31, ty = threadIdx.x >> 5;
#pragma unroll
  for (int j = 0; j < 4; ++j)
    tile[ty + j * 8][tx] = in[(long long)(r0 + ty + j * 8) * ldin + (c0 + tx)];
  __syncthreads();
#pragma unroll
  for (int j = 0; j < 4; ++j)
    out[(long long)(c0 + ty + j * 8) * R + (r0 + tx)] = tile[tx][ty + j * 8];
}

// ---------------------------------------------------------------------------
// Elementwise kernels (compute f32, store f16 where feeding WMMA)
// ---------------------------------------------------------------------------
__global__ __launch_bounds__(256) void cvt_f32_f16_kernel(
    const float* __restrict__ in, f16* __restrict__ out, long long n)
{
  const long long i = (long long)blockIdx.x * 256 + threadIdx.x;
  if (i < n) out[i] = (f16)in[i];
}

__global__ __launch_bounds__(256) void rmsnorm_kernel(
    const float* __restrict__ x, const float* __restrict__ w,
    f16* __restrict__ y, int d)
{
  __shared__ float red[256];
  const int row = blockIdx.x;
  const float* xr = x + (long long)row * d;
  float s = 0.f;
  for (int i = threadIdx.x; i < d; i += 256) { float v = xr[i]; s += v * v; }
  red[threadIdx.x] = s;
  __syncthreads();
  for (int st = 128; st > 0; st >>= 1) {
    if (threadIdx.x < st) red[threadIdx.x] += red[threadIdx.x + st];
    __syncthreads();
  }
  const float inv = rsqrtf(red[0] / (float)d + 1e-6f);
  f16* yr = y + (long long)row * d;
  for (int i = threadIdx.x; i < d; i += 256) yr[i] = (f16)(xr[i] * inv * w[i]);
}

__global__ __launch_bounds__(256) void rope_kernel(
    const f16* __restrict__ xin, f16* __restrict__ xout,
    const float* __restrict__ ct, const float* __restrict__ st, int nH)
{
  const long long idx = (long long)blockIdx.x * 256 + threadIdx.x;
  const int cols = nH * HD;
  const int srow = (int)(idx / cols);
  const int col  = (int)(idx % cols);
  const int d    = col & (HD - 1);
  const float v  = (float)xin[idx];
  const float o  = (d < HD / 2) ? -(float)xin[idx + HD / 2] : (float)xin[idx - HD / 2];
  xout[idx] = (f16)(v * ct[srow * HD + d] + o * st[srow * HD + d]);
}

__global__ __launch_bounds__(256) void softmax_kernel(
    f16* __restrict__ sc, const float* __restrict__ mask, float scale, int s)
{
  __shared__ float red[256];
  const int row = blockIdx.x;
  const int h   = blockIdx.y;
  f16* p = sc + ((long long)h * s + row) * s;
  const float* mr = mask + (long long)row * s;

  float mx = -3.4e38f;
  for (int j = threadIdx.x; j < s; j += 256)
    mx = fmaxf(mx, (float)p[j] * scale + mr[j]);
  red[threadIdx.x] = mx;
  __syncthreads();
  for (int st = 128; st > 0; st >>= 1) {
    if (threadIdx.x < st) red[threadIdx.x] = fmaxf(red[threadIdx.x], red[threadIdx.x + st]);
    __syncthreads();
  }
  mx = red[0];
  __syncthreads();

  float sum = 0.f;
  for (int j = threadIdx.x; j < s; j += 256) {
    const float e = expf((float)p[j] * scale + mr[j] - mx);
    p[j] = (f16)e;
    sum += e;
  }
  red[threadIdx.x] = sum;
  __syncthreads();
  for (int st = 128; st > 0; st >>= 1) {
    if (threadIdx.x < st) red[threadIdx.x] += red[threadIdx.x + st];
    __syncthreads();
  }
  const float inv = 1.f / red[0];
  for (int j = threadIdx.x; j < s; j += 256) p[j] = (f16)((float)p[j] * inv);
}

__global__ __launch_bounds__(256) void gelu_mul_kernel(
    const f16* __restrict__ g, const f16* __restrict__ u,
    f16* __restrict__ o, long long n)
{
  const long long i = (long long)blockIdx.x * 256 + threadIdx.x;
  if (i >= n) return;
  const float x = (float)g[i];
  const float t = tanhf(0.7978845608028654f * (x + 0.044715f * x * x * x));
  o[i] = (f16)(0.5f * x * (1.f + t) * (float)u[i]);
}

__global__ __launch_bounds__(256) void copy_kernel(
    const float* __restrict__ a, float* __restrict__ b, long long n)
{
  const long long i = (long long)blockIdx.x * 256 + threadIdx.x;
  if (i < n) b[i] = a[i];
}

// ---------------------------------------------------------------------------
// Host orchestration
// ---------------------------------------------------------------------------
extern "C" void kernel_launch(void* const* d_in, const int* in_sizes, int n_in,
                              void* d_out, int out_size, void* d_ws, size_t ws_size,
                              hipStream_t stream) {
  (void)in_sizes; (void)n_in; (void)out_size; (void)ws_size;
  const float* hidden     = (const float*)d_in[0];
  const float* mask_full  = (const float*)d_in[1];
  const float* mask_slide = (const float*)d_in[2];
  const float* plc        = (const float*)d_in[4];
  const float* cos_s      = (const float*)d_in[5];
  const float* sin_s      = (const float*)d_in[6];
  const float* cos_f      = (const float*)d_in[7];
  const float* sin_f      = (const float*)d_in[8];
  const float* inj_k[2]   = {(const float*)d_in[9],  (const float*)d_in[11]};
  const float* inj_v[2]   = {(const float*)d_in[10], (const float*)d_in[12]};
  const float* Wq  = (const float*)d_in[13];
  const float* Wk  = (const float*)d_in[14];
  const float* Wv  = (const float*)d_in[15];
  const float* Wo  = (const float*)d_in[16];
  const float* ln1 = (const float*)d_in[17];
  const float* ln2 = (const float*)d_in[18];
  const float* Wg  = (const float*)d_in[19];
  const float* Wu  = (const float*)d_in[20];
  const float* Wd  = (const float*)d_in[21];
  const float* Wpl = (const float*)d_in[22];
  float* out = (float*)d_out;

  // ---- workspace carve-up ----
  float* h = (float*)d_ws;                            // S*D f32 residual stream
  f16* fp = (f16*)(h + (long long)S * D);
  auto take = [&](long long n) { f16* p = fp; fp += n; return p; };
  f16* x16   = take((long long)S * D);
  f16* q16   = take((long long)S * H * HD);
  f16* q2    = take((long long)S * H * HD);
  f16* k16   = take((long long)S * KVH * HD);
  f16* k2    = take((long long)S * KVH * HD);
  f16* v16   = take((long long)S * KVH * HD);
  f16* vT    = take((long long)KVH * HD * S);
  f16* o16   = take((long long)S * H * HD);
  f16* g16   = take((long long)S * F);
  f16* u16   = take((long long)S * F);
  f16* sc16  = take((long long)H * S * S);
  f16* plc16 = take((long long)S * L * PLD);
  f16* wq_t  = take((long long)(H * HD) * D);
  f16* wk_t  = take((long long)(KVH * HD) * D);
  f16* wv_t  = take((long long)(KVH * HD) * D);
  f16* wo_t  = take((long long)D * (H * HD));
  f16* wg_t  = take((long long)F * D);
  f16* wu_t  = take((long long)F * D);
  f16* wd_t  = take((long long)D * F);
  f16* wpl_t = take((long long)D * PLD);

  auto gemm = [&](const f16* A, const f16* B, const float* Cin,
                  float* Cf, f16* Ch,
                  int M, int N, int K, int lda, int ldb, int ldc, int ldcin,
                  long long aHS, long long bHS, long long cHS, long long cinHS,
                  int bDiv, int nz) {
    dim3 grid(N / 256, M / 128, nz), block(256);
    wmma_gemm_f16_kernel<<<grid, block, 0, stream>>>(
        A, B, Cin, Cf, Ch, K, lda, ldb, ldc, ldcin,
        aHS, bHS, cHS, cinHS, bDiv);
  };
  auto trans32 = [&](const float* in, f16* o2, int R, int C, int ldin,
                     long long inBS, long long outBS, int nz) {
    transpose_cvt_f32_kernel<<<dim3(C / 32, R / 32, nz), 256, 0, stream>>>(
        in, o2, R, C, ldin, inBS, outBS);
  };

  const long long nSD = (long long)S * D;
  copy_kernel<<<(int)(nSD / 256), 256, 0, stream>>>(hidden, h, nSD);
  cvt_f32_f16_kernel<<<(int)((long long)S * L * PLD / 256), 256, 0, stream>>>(
      plc, plc16, (long long)S * L * PLD);

  const float scale = 0.0625f;  // HD^-0.5

  for (int l = 0; l < L; ++l) {
    const int layer = 12 + l;
    const bool full = (layer % 6) == 5;
    const float* ct = full ? cos_f : cos_s;
    const float* st = full ? sin_f : sin_s;
    const float* mask = full ? mask_full : mask_slide;
    const bool inj = (layer == 13 || layer == 14);

    // ---- weight transpose-conversions for this layer: Wt[n][k] = W[k][n] ----
    trans32(Wq + (long long)l * D * (H * HD), wq_t, D, H * HD, H * HD, 0, 0, 1);
    trans32(Wo + (long long)l * (H * HD) * D, wo_t, H * HD, D, D, 0, 0, 1);
    trans32(Wg + (long long)l * D * F, wg_t, D, F, F, 0, 0, 1);
    trans32(Wu + (long long)l * D * F, wu_t, D, F, F, 0, 0, 1);
    trans32(Wd + (long long)l * F * D, wd_t, F, D, D, 0, 0, 1);
    trans32(Wpl + (long long)l * PLD * D, wpl_t, PLD, D, D, 0, 0, 1);
    if (!inj) {
      trans32(Wk + (long long)l * D * (KVH * HD), wk_t, D, KVH * HD, KVH * HD, 0, 0, 1);
      trans32(Wv + (long long)l * D * (KVH * HD), wv_t, D, KVH * HD, KVH * HD, 0, 0, 1);
    }

    // ---- attention ----
    rmsnorm_kernel<<<S, 256, 0, stream>>>(h, ln1 + (long long)l * D, x16, D);
    gemm(x16, wq_t, nullptr, nullptr, q16,
         S, H * HD, D, D, D, H * HD, 0, 0, 0, 0, 0, 1, 1);
    rope_kernel<<<(int)((long long)S * H * HD / 256), 256, 0, stream>>>(q16, q2, ct, st, H);

    int kldb; long long kHS;
    if (inj) {
      // injected pre-RoPE'd KV: [KVH, S, HD]
      cvt_f32_f16_kernel<<<(int)((long long)KVH * S * HD / 256), 256, 0, stream>>>(
          inj_k[layer - 13], k2, (long long)KVH * S * HD);
      trans32(inj_v[layer - 13], vT, S, HD, HD, (long long)S * HD, (long long)HD * S, KVH);
      kldb = HD; kHS = (long long)S * HD;
    } else {
      gemm(x16, wk_t, nullptr, nullptr, k16,
           S, KVH * HD, D, D, D, KVH * HD, 0, 0, 0, 0, 0, 1, 1);
      gemm(x16, wv_t, nullptr, nullptr, v16,
           S, KVH * HD, D, D, D, KVH * HD, 0, 0, 0, 0, 0, 1, 1);
      rope_kernel<<<(int)((long long)S * KVH * HD / 256), 256, 0, stream>>>(k16, k2, ct, st, KVH);
      transpose_cvt_f16_kernel<<<dim3(HD / 32, S / 32, KVH), 256, 0, stream>>>(
          v16, vT, S, HD, KVH * HD, (long long)HD, (long long)HD * S);
      kldb = KVH * HD; kHS = (long long)HD;
    }

    // scores[h] = q[h] * k[h/4]^T   (M=S, N=S, K=HD)
    gemm(q2, k2, nullptr, nullptr, sc16,
         S, S, HD, H * HD, kldb, S, 0,
         (long long)HD, kHS, (long long)S * S, 0, 4, H);
    softmax_kernel<<<dim3(S, H), 256, 0, stream>>>(sc16, mask, scale, S);

    // o[h] = attn[h] * vT[h/4]^T    (M=S, N=HD, K=S)
    gemm(sc16, vT, nullptr, nullptr, o16,
         S, HD, S, S, S, H * HD, 0,
         (long long)S * S, (long long)HD * S, (long long)HD, 0, 4, H);

    // h = h + o * Wo
    gemm(o16, wo_t, h, h, nullptr,
         S, D, H * HD, H * HD, H * HD, D, D, 0, 0, 0, 0, 1, 1);

    // ---- mlp ----
    rmsnorm_kernel<<<S, 256, 0, stream>>>(h, ln2 + (long long)l * D, x16, D);
    gemm(x16, wg_t, nullptr, nullptr, g16, S, F, D, D, D, F, 0, 0, 0, 0, 0, 1, 1);
    gemm(x16, wu_t, nullptr, nullptr, u16, S, F, D, D, D, F, 0, 0, 0, 0, 0, 1, 1);
    gelu_mul_kernel<<<(int)((long long)S * F / 256), 256, 0, stream>>>(
        g16, u16, g16, (long long)S * F);
    gemm(g16, wd_t, h, h, nullptr, S, D, F, F, F, D, D, 0, 0, 0, 0, 1, 1);

    // ---- per-layer injected embedding: h += plc[:, :, l, :] * Wpl[l] ----
    gemm(plc16 + (long long)l * PLD, wpl_t, h, h, nullptr,
         S, D, PLD, L * PLD, PLD, D, D, 0, 0, 0, 0, 1, 1);

    if (layer == 17) {
      copy_kernel<<<(int)(nSD / 256), 256, 0, stream>>>(h, out + nSD, nSD);
    }
  }

  copy_kernel<<<(int)(nSD / 256), 256, 0, stream>>>(h, out, nSD);
}
